// LDEModule_67860483277127
// MI455X (gfx1250) — matrix operand order repaired and checked by reading, not compiled, not run on hardware
//
#include <hip/hip_runtime.h>
#include <hip/hip_fp16.h>

// ---------------------------------------------------------------------------
// LDE module, MI455X (gfx1250) fused implementation (f16 WMMA, f32 acc/state).
//  B=4096, T=16, D_IN=2048, H=1024, MID=100, L=25
//  out0 = x_last @ W_down + b_down                              (WMMA)
//  coef = SIGMA*sigmoid(relu(out0@W_diff1+b1)@W_diff2+b2)*DT^(2/3)   (WMMA head)
//  25x:  out += relu(out @ W_drift + b_drift)*DT + coef*noise[l]     (WMMA)
// Each batch row is independent in the scan -> one WG owns a 16-row tile and
// runs the whole recurrence with LDS-resident f32 state; packed f16 weights
// stay L2-resident; per-step noise tile (64KB contiguous) is prefetched under
// the WMMA K-loop via global_prefetch_b8.
// ---------------------------------------------------------------------------

#define BB   4096
#define TT   16
#define DIN  2048
#define HH   1024
#define MIDD 100
#define LL   25
#define SIGMA_C 0.5f
#define DT_C (1.0f / 25.0f)
#define MTILES 7          // ceil(MID/16)

typedef __attribute__((ext_vector_type(16))) _Float16 v16h;
typedef __attribute__((ext_vector_type(8)))  _Float16 v8h;
typedef __attribute__((ext_vector_type(8)))  float    v8f;

// LDS row strides (elements), padded for conflict-free 16-row column access
#define SOF 1028   // f32 state stride  (1028 % 64 == 4 -> distinct banks per row)
#define SOH 1032   // f16 state stride (516 dwords, 16B-aligned rows)

// ---- fragment loaders -----------------------------------------------------

// A-frag (16x32 f16) converted from a global f32 row.
// p -> &row[kc*32 + (lane>>4)*8]; elems 0..7 -> K base+0..7, 8..15 -> +16..23
__device__ __forceinline__ v16h load_a_global_f32(const float* __restrict__ p) {
  float4 q0 = *(const float4*)(p + 0);
  float4 q1 = *(const float4*)(p + 4);
  float4 q2 = *(const float4*)(p + 16);
  float4 q3 = *(const float4*)(p + 20);
  v16h a;
  a[0]=(_Float16)q0.x; a[1]=(_Float16)q0.y; a[2]=(_Float16)q0.z; a[3]=(_Float16)q0.w;
  a[4]=(_Float16)q1.x; a[5]=(_Float16)q1.y; a[6]=(_Float16)q1.z; a[7]=(_Float16)q1.w;
  a[8]=(_Float16)q2.x; a[9]=(_Float16)q2.y; a[10]=(_Float16)q2.z; a[11]=(_Float16)q2.w;
  a[12]=(_Float16)q3.x; a[13]=(_Float16)q3.y; a[14]=(_Float16)q3.z; a[15]=(_Float16)q3.w;
  return a;
}

// A-frag from LDS f16 row (two 16B chunks: K base+0..7 and base+16..23)
__device__ __forceinline__ v16h load_a_lds(const _Float16* p) {
  v8h lo = *(const v8h*)(p);
  v8h hi = *(const v8h*)(p + 16);
  v16h a;
#pragma unroll
  for (int e = 0; e < 8; ++e) { a[e] = lo[e]; a[e + 8] = hi[e]; }
  return a;
}

// B-frag from packed layout: frag (kc, nt) = 32 lanes x 16 contiguous f16
__device__ __forceinline__ v16h load_b_pk(const _Float16* __restrict__ pk,
                                          int kc, int nt, int lane, int ntiles) {
  return *(const v16h*)(pk + (((size_t)(kc * ntiles + nt) * 32 + lane) << 4));
}

// ---- kernel 0: pack KxN f32 matrix into B-fragment f16 layout -------------
// packed[((kt*ntiles+nt)*32+lane)*16+e] =
//     f16( W[kt*32 + (lane>>4)*8 + e + (e&8)][nt*16 + (lane&15)] ), zero if col >= N
__global__ void pack_b_f16(const float* __restrict__ W, _Float16* __restrict__ pk,
                           int K, int N, int ntiles) {
  int idx = blockIdx.x * blockDim.x + threadIdx.x;
  int total = (K >> 5) * ntiles * 32;
  if (idx >= total) return;
  int lane = idx & 31;
  int t    = idx >> 5;
  int nt   = t % ntiles;
  int kt   = t / ntiles;
  int col  = nt * 16 + (lane & 15);
  int kb   = kt * 32 + ((lane >> 4) << 3);
  _Float16* dst = pk + ((size_t)idx << 4);
  if (col < N) {
    const float* src = W + (size_t)kb * N + col;
#pragma unroll
    for (int e = 0; e < 16; ++e) dst[e] = (_Float16)src[(size_t)(e + (e & 8)) * N];
  } else {
#pragma unroll
    for (int e = 0; e < 16; ++e) dst[e] = (_Float16)0.f;
  }
}

// ---- kernel 1: fused down-GEMM + WMMA diffusion head + 25-step scan -------
__global__ __launch_bounds__(256)
void lde_fused(const float* __restrict__ x_seq,
               const float* __restrict__ noise,
               const _Float16* __restrict__ Wdown_pk,
               const float* __restrict__ b_down,
               const _Float16* __restrict__ Wdrift_pk,
               const float* __restrict__ b_drift,
               const _Float16* __restrict__ Wdiff1_pk,
               const float* __restrict__ b_diff1,
               const float* __restrict__ W_diff2,
               const float* __restrict__ b_diff2,
               float* __restrict__ out) {
  __shared__ float    sOut[16 * SOF];   // f32 state  (16 x 1024, padded)
  __shared__ _Float16 sOutH[16 * SOH];  // f16 copy for A-fragments
  __shared__ float    sMid[16 * 104];   // relu(out0@Wd1+b1)*Wd2 partials
  __shared__ float    sCoef[16];        // per-row noise coefficient

  const int tid  = threadIdx.x;
  const int wave = tid >> 5;
  const int lane = tid & 31;
  const int row0 = blockIdx.x * 16;

  // fragment coordinates for this lane
  const int am    = lane & 15;         // A-frag row
  const int akb   = (lane >> 4) * 8;   // A-frag K sub-base
  const int cn    = lane & 15;         // C-frag column
  const int rbase = (lane >> 4) * 8;   // C-frag row base

  const _Float16* arow = sOutH + am * SOH;

  // ---------------- Phase A: out0 = x_last @ W_down + b_down ----------------
  {
    const float* xrow = x_seq + ((size_t)(row0 + am) * TT + (TT - 1)) * DIN;
    v8f acc[8] = {};
    for (int kc = 0; kc < DIN / 32; ++kc) {
      v16h a = load_a_global_f32(xrow + kc * 32 + akb);
#pragma unroll
      for (int j = 0; j < 8; ++j) {
        v16h b = load_b_pk(Wdown_pk, kc, wave * 8 + j, lane, HH / 16);
        acc[j] = __builtin_amdgcn_wmma_f32_16x16x32_f16(
            false, a, false, b, (short)0, acc[j], false, false);
      }
    }
#pragma unroll
    for (int j = 0; j < 8; ++j) {
      int nn = (wave * 8 + j) * 16 + cn;
      float bd = b_down[nn];
#pragma unroll
      for (int r = 0; r < 8; ++r) {
        int m = rbase + r;
        float v = acc[j][r] + bd;
        sOut[m * SOF + nn]  = v;
        sOutH[m * SOH + nn] = (_Float16)v;
      }
    }
  }
  __syncthreads();

  // ------- Phase B: diffusion head (WMMA 16x1024x112) -> per-row coef -------
  if (wave < MTILES) {
    v8f acc = {};
    for (int kc = 0; kc < HH / 32; ++kc) {
      v16h a = load_a_lds(arow + kc * 32 + akb);
      v16h b = load_b_pk(Wdiff1_pk, kc, wave, lane, MTILES);
      acc = __builtin_amdgcn_wmma_f32_16x16x32_f16(
          false, a, false, b, (short)0, acc, false, false);
    }
    int n = wave * 16 + cn;
    if (n < MIDD) {
      float b1 = b_diff1[n], w2 = W_diff2[n];
#pragma unroll
      for (int r = 0; r < 8; ++r) {
        int m = rbase + r;
        sMid[m * 104 + n] = fmaxf(acc[r] + b1, 0.f) * w2;
      }
    }
  }
  __syncthreads();
  if (tid < 16) {
    float s = b_diff2[0];
    for (int mid = 0; mid < MIDD; ++mid) s += sMid[tid * 104 + mid];
    float sig = 1.f / (1.f + expf(-s));
    sCoef[tid] = SIGMA_C * sig * powf(DT_C, 1.0f / 1.5f);
  }
  __syncthreads();

  // ---------------- Phase C: 25-step scan -----------------------------------
  for (int l = 0; l < LL; ++l) {
    // Prefetch this step's contiguous 64KB noise tile under the WMMA K-loop.
    const float* nbase = noise + ((size_t)l * BB + row0) * HH;
    for (int off = tid * 32; off < 16 * HH; off += 256 * 32)
      __builtin_prefetch(nbase + off, 0, 3);

    v8f acc[8] = {};
    for (int kc = 0; kc < HH / 32; ++kc) {
      v16h a = load_a_lds(arow + kc * 32 + akb);
#pragma unroll
      for (int j = 0; j < 8; ++j) {
        v16h b = load_b_pk(Wdrift_pk, kc, wave * 8 + j, lane, HH / 16);
        acc[j] = __builtin_amdgcn_wmma_f32_16x16x32_f16(
            false, a, false, b, (short)0, acc[j], false, false);
      }
    }
    __syncthreads();  // everyone done reading sOutH for this step
#pragma unroll
    for (int j = 0; j < 8; ++j) {
      int nn = (wave * 8 + j) * 16 + cn;
      float bd = b_drift[nn];
      const float* nz = nbase + nn;
#pragma unroll
      for (int r = 0; r < 8; ++r) {
        int m = rbase + r;
        float drift = fmaxf(acc[j][r] + bd, 0.f);
        float nv = sOut[m * SOF + nn] + drift * DT_C +
                   sCoef[m] * nz[(size_t)m * HH];
        sOut[m * SOF + nn]  = nv;
        sOutH[m * SOH + nn] = (_Float16)nv;
      }
    }
    __syncthreads();
  }

  // ---------------- write result (coalesced) --------------------------------
  for (int t = tid; t < 16 * HH; t += 256) {
    int m = t >> 10, n = t & (HH - 1);
    out[(size_t)(row0 + m) * HH + n] = sOut[m * SOF + n];
  }
}

// ---------------------------------------------------------------------------
extern "C" void kernel_launch(void* const* d_in, const int* in_sizes, int n_in,
                              void* d_out, int out_size, void* d_ws, size_t ws_size,
                              hipStream_t stream) {
  (void)in_sizes; (void)n_in; (void)out_size; (void)ws_size;
  const float* x_seq   = (const float*)d_in[0];
  const float* noise   = (const float*)d_in[1];
  const float* W_down  = (const float*)d_in[2];
  const float* b_down  = (const float*)d_in[3];
  const float* W_drift = (const float*)d_in[4];
  const float* b_drift = (const float*)d_in[5];
  const float* W_diff1 = (const float*)d_in[6];
  const float* b_diff1 = (const float*)d_in[7];
  const float* W_diff2 = (const float*)d_in[8];
  const float* b_diff2 = (const float*)d_in[9];
  float* out = (float*)d_out;

  _Float16* Wdown_pk  = (_Float16*)d_ws;                     // 2048*1024 f16 = 4 MB
  _Float16* Wdrift_pk = Wdown_pk + (size_t)DIN * HH;         // 1024*1024 f16 = 2 MB
  _Float16* Wdiff1_pk = Wdrift_pk + (size_t)HH * HH;         // 32*7*512   f16 = 224 KB

  int tot_down  = (DIN / 32) * (HH / 16) * 32;    // 131072
  int tot_drift = (HH / 32) * (HH / 16) * 32;     //  65536
  int tot_diff1 = (HH / 32) * MTILES * 32;        //   7168
  pack_b_f16<<<tot_down / 256, 256, 0, stream>>>(W_down, Wdown_pk, DIN, HH, HH / 16);
  pack_b_f16<<<tot_drift / 256, 256, 0, stream>>>(W_drift, Wdrift_pk, HH, HH, HH / 16);
  pack_b_f16<<<(tot_diff1 + 255) / 256, 256, 0, stream>>>(W_diff1, Wdiff1_pk, HH, MIDD, MTILES);

  lde_fused<<<BB / 16, 256, 0, stream>>>(
      x_seq, noise, Wdown_pk, b_down, Wdrift_pk, b_drift,
      Wdiff1_pk, b_diff1, W_diff2, b_diff2, out);
}
